// CustomAttention_1675037245778
// MI455X (gfx1250) — matrix operand order repaired and checked by reading, not compiled, not run on hardware
//
#include <hip/hip_runtime.h>

typedef _Float16 half_t;
typedef __attribute__((ext_vector_type(16))) _Float16 v16h;
typedef __attribute__((ext_vector_type(8)))  _Float16 v8h;
typedef __attribute__((ext_vector_type(8)))  float    v8f;
typedef unsigned int v4u __attribute__((ext_vector_type(4)));
typedef unsigned int v8u __attribute__((ext_vector_type(8)));

#define WMMA(a,b,c) __builtin_amdgcn_wmma_f32_16x16x32_f16(false,(a),false,(b),(short)0,(c),false,false)

static constexpr int SEQ = 2048;
static constexpr int CH  = 1024;
static constexpr int NH  = 16;
static constexpr int HD  = 64;

union H16 { v16h v; v8h h[2]; };

// A-matrix 16x32 f16 tile (row-major src, leading dim ld).
// Lane L: row = L&15; hi = L>>4; halves 0..7 <-> k = hi*8 .. hi*8+7,
// halves 8..15 <-> k = hi*8+16 .. hi*8+23 (per ISA 16-bit A layout).
__device__ __forceinline__ v16h load_a16x32(const half_t* src, int ld) {
  int lane = threadIdx.x & 31;
  int row = lane & 15, hi = lane >> 4;
  const half_t* p = src + row * ld + hi * 8;
  H16 u;
  u.h[0] = *(const v8h*)(p);
  u.h[1] = *(const v8h*)(p + 16);
  return u.v;
}

// B-matrix 32x16 f16 tile. Lane L: col(n) = L&15; halves 0..7 <-> k = hi*8 .. hi*8+7,
// halves 8..15 <-> k = hi*8+16 .. hi*8+23 (matches A layout).
__device__ __forceinline__ v16h load_b32x16(const half_t* src, int ld) {
  int lane = threadIdx.x & 31;
  int col = lane & 15, hi = lane >> 4;
  const half_t* p = src + col * ld + hi * 8;
  H16 u;
  u.h[0] = *(const v8h*)(p);
  u.h[1] = *(const v8h*)(p + 16);
  return u.v;
}

// Tensor Data Mover: 2D tile load Global->LDS, f16 elements.
// D# built per CDNA5 ISA 8.3/8.4: group0 = {count=1, lds_addr, global_addr(57b), type=2},
// group1 = {data_size=2B, tensor dims == tile dims, dim0 stride}.
// Descriptor values are block-uniform -> SGPRs; issued by one wave, tracked by TENSORcnt.
__device__ __forceinline__ void tdm_load_2d(const half_t* g, unsigned lds_off,
                                            unsigned row_elems, unsigned rows,
                                            unsigned row_stride_elems) {
  unsigned long long ga = (unsigned long long)(uintptr_t)g;
  v4u g0;
  g0.x = 1u;                                                  // count=1, user mode
  g0.y = lds_off;                                             // LDS byte address
  g0.z = (unsigned)ga;                                        // global_addr[31:0]
  g0.w = (unsigned)((ga >> 32) & 0x1FFFFFFu) | (2u << 30);    // global_addr[56:32] | type=2
  v8u g1;
  g1.s0 = 1u << 16;                                           // data_size = 2 bytes
  g1.s1 = (row_elems & 0xFFFFu) << 16;                        // tensor_dim0[15:0]
  g1.s2 = (row_elems >> 16) | ((rows & 0xFFFFu) << 16);       // tensor_dim0[31:16] | tensor_dim1[15:0]
  g1.s3 = (rows >> 16) | ((row_elems & 0xFFFFu) << 16);       // tensor_dim1[31:16] | tile_dim0
  g1.s4 = rows & 0xFFFFu;                                     // tile_dim1 (tile_dim2 = 0)
  g1.s5 = row_stride_elems;                                   // tensor_dim0_stride[31:0]
  g1.s6 = 0u;                                                 // stride hi / dim1_stride lo
  g1.s7 = 0u;
  asm volatile("tensor_load_to_lds %0, %1" :: "s"(g0), "s"(g1) : "memory");
}

__global__ void cvt_f32_to_f16(const float* __restrict__ s, half_t* __restrict__ d, int n) {
  int i = blockIdx.x * blockDim.x + threadIdx.x;
  if (i < n) d[i] = (half_t)s[i];
}

// qkv = x @ w_qkv^T ; scatter into q [B,H,N,D], k [B,H,N,D], v^T [B,H,D,N] (f16)
__global__ __launch_bounds__(256) void gemm_qkv(
    const half_t* __restrict__ xh, const half_t* __restrict__ wh,
    half_t* __restrict__ qh, half_t* __restrict__ kh, half_t* __restrict__ vth)
{
  int wave = threadIdx.x >> 5;
  int gw = blockIdx.x * 8 + wave;        // 12288 wave-tiles
  int m0 = (gw & 255) * 16;              // 256 row tiles (M = 4096)
  int n0 = (gw >> 8) * 64;               // 48 col tiles  (N = 3072)
  int lane = threadIdx.x & 31;
  int cl = lane & 15, hi = lane >> 4;

  v8f acc[4] = {};
  const half_t* A = xh + (size_t)m0 * CH;
#pragma unroll 2
  for (int k0 = 0; k0 < CH; k0 += 32) {
    // stage A + all 4 B tiles first so loads clause together and the
    // 4 WMMAs issue back-to-back behind a single s_wait_loadcnt
    v16h a = load_a16x32(A + k0, CH);
    v16h b[4];
#pragma unroll
    for (int t = 0; t < 4; ++t)
      b[t] = load_b32x16(wh + (size_t)(n0 + t * 16) * CH + k0, CH);
#pragma unroll
    for (int t = 0; t < 4; ++t)
      acc[t] = WMMA(a, b[t], acc[t]);
  }
#pragma unroll
  for (int t = 0; t < 4; ++t)
#pragma unroll
    for (int r = 0; r < 8; ++r) {
      float val = acc[t][r];
      int m = m0 + r + 8 * hi;           // row in [0, 4096)
      int col = n0 + t * 16 + cl;        // col in [0, 3072)
      int three = col >> 10;
      int rem = col & 1023;
      int h = rem >> 6, d = rem & 63;
      int bb = m >> 11, n = m & 2047;
      if (three == 0)
        qh[(((size_t)bb * NH + h) * SEQ + n) * HD + d] = (half_t)val;
      else if (three == 1)
        kh[(((size_t)bb * NH + h) * SEQ + n) * HD + d] = (half_t)val;
      else
        vth[(((size_t)bb * NH + h) * HD + d) * SEQ + n] = (half_t)val;
    }
}

// Flash attention with multiplicative mask: per wave a 16-row Q tile; K / V^T
// blocks streamed into double-buffered LDS by the Tensor Data Mover (wave 0
// issues the DMA; TENSORcnt + barrier publish it to all 8 waves of the block).
__global__ __launch_bounds__(256) void flash_attn(
    const half_t* __restrict__ qh, const half_t* __restrict__ kh,
    const half_t* __restrict__ vth, const float* __restrict__ mask,
    half_t* __restrict__ oh)
{
  __shared__ __align__(16) half_t ldsK[2][32 * HD];   // K block  [key][d]
  __shared__ __align__(16) half_t ldsV[2][HD * 32];   // V^T block [d][key]
  __shared__ __align__(16) half_t ldsP[8][16 * 32];   // per-wave P scratch

  int wave = threadIdx.x >> 5;
  int lane = threadIdx.x & 31;
  int cl = lane & 15, hi = lane >> 4;
  int bh = blockIdx.x >> 4;              // (b*16 + h), 32 values
  int qblk = blockIdx.x & 15;
  int q0 = qblk * 128 + wave * 16;
  int b = bh >> 4, h = bh & 15;

  const half_t* Q  = qh + ((size_t)bh * SEQ + q0) * HD;
  const half_t* Kg = kh + (size_t)bh * SEQ * HD;      // [SEQ][64]
  const half_t* Vg = vth + (size_t)bh * HD * SEQ;     // [64][SEQ]

  v16h qa0 = load_a16x32(Q, HD);
  v16h qa1 = load_a16x32(Q + 32, HD);

  v8f o[4] = {};
  float mrow[8], lrow[8];
#pragma unroll
  for (int r = 0; r < 8; ++r) { mrow[r] = -1e30f; lrow[r] = 0.f; }

  const float scale = 0.125f;            // 1/sqrt(64)

  // prologue: TDM-load tile 0 into buffer 0
  if (wave == 0) {
    tdm_load_2d(Kg, (unsigned)(uintptr_t)&ldsK[0][0], HD, 32, HD);
    tdm_load_2d(Vg, (unsigned)(uintptr_t)&ldsV[0][0], 32, HD, SEQ);
  }

  for (int kb = 0; kb < SEQ; kb += 32) {
    int buf = (kb >> 5) & 1;
    if (wave == 0) __builtin_amdgcn_s_wait_tensorcnt(0);  // current tile landed
    __syncthreads();                                      // publish LDS to all waves
    if (wave == 0 && kb + 32 < SEQ) {                     // overlap next DMA with compute
      tdm_load_2d(Kg + (size_t)(kb + 32) * HD,
                  (unsigned)(uintptr_t)&ldsK[buf ^ 1][0], HD, 32, HD);
      tdm_load_2d(Vg + (kb + 32),
                  (unsigned)(uintptr_t)&ldsV[buf ^ 1][0], 32, HD, SEQ);
    }
    if (kb + 32 < SEQ)
      __builtin_prefetch(mask + (size_t)(q0 + cl) * SEQ + kb + 32, 0, 0);

    // S = Q K^T : two 16x16 f32 blocks (keys kb..+15 and kb+16..+31)
    v16h kt[4];
#pragma unroll
    for (int t = 0; t < 2; ++t) {
      kt[2 * t]     = load_b32x16(&ldsK[buf][t * 16 * HD], HD);
      kt[2 * t + 1] = load_b32x16(&ldsK[buf][t * 16 * HD + 32], HD);
    }
    v8f s[2];
#pragma unroll
    for (int t = 0; t < 2; ++t) {
      v8f a{};
      a = WMMA(qa0, kt[2 * t], a);
      a = WMMA(qa1, kt[2 * t + 1], a);
      s[t] = a;
    }

    // multiplicative mask + online softmax (rows M = r + 8*hi, cols across lanes)
    float alpha[8];
    half_t* P = ldsP[wave];
#pragma unroll
    for (int r = 0; r < 8; ++r) {
      int row = q0 + r + 8 * hi;
      const float* mk = mask + (size_t)row * SEQ + kb;
      float s0 = s[0][r] * scale * mk[cl];
      float s1 = s[1][r] * scale * mk[16 + cl];
      float mx = fmaxf(s0, s1);
#pragma unroll
      for (int off = 8; off; off >>= 1) mx = fmaxf(mx, __shfl_xor(mx, off, 16));
      float mn = fmaxf(mrow[r], mx);
      float al = __expf(mrow[r] - mn);
      float p0 = __expf(s0 - mn);
      float p1 = __expf(s1 - mn);
      float rs = p0 + p1;
#pragma unroll
      for (int off = 8; off; off >>= 1) rs += __shfl_xor(rs, off, 16);
      lrow[r] = lrow[r] * al + rs;
      mrow[r] = mn;
      alpha[r] = al;
      int prow = r + 8 * hi;
      P[prow * 32 + cl]      = (half_t)p0;
      P[prow * 32 + 16 + cl] = (half_t)p1;
    }
    asm volatile("s_wait_dscnt 0" ::: "memory");   // cross-lane LDS visibility (wave-local scratch)

    // rescale O by per-row alpha, then O += P(16x32) x V(32x64)
#pragma unroll
    for (int t = 0; t < 4; ++t)
#pragma unroll
      for (int r = 0; r < 8; ++r)
        o[t][r] *= alpha[r];

    v16h pa = load_a16x32(P, 32);
    v16h vb[4];
#pragma unroll
    for (int t = 0; t < 4; ++t)
      vb[t] = load_b32x16(&ldsV[buf][t * 16 * 32], 32);
#pragma unroll
    for (int t = 0; t < 4; ++t)
      o[t] = WMMA(pa, vb[t], o[t]);
  }

  // normalize and store to [B, N, C] (C = h*64 + d) as f16 for the proj gemm
#pragma unroll
  for (int t = 0; t < 4; ++t)
#pragma unroll
    for (int r = 0; r < 8; ++r) {
      int row = q0 + r + 8 * hi;
      int d = t * 16 + cl;
      float val = o[t][r] / lrow[r];
      oh[((size_t)b * SEQ + row) * CH + h * HD + d] = (half_t)val;
    }
}

// out = attn_out @ w_proj^T + b_proj  (f32 output)
__global__ __launch_bounds__(256) void gemm_proj(
    const half_t* __restrict__ ah, const half_t* __restrict__ wh,
    const float* __restrict__ bias, float* __restrict__ out)
{
  int wave = threadIdx.x >> 5;
  int gw = blockIdx.x * 8 + wave;        // 4096 wave-tiles
  int m0 = (gw & 255) * 16;              // M = 4096
  int n0 = (gw >> 8) * 64;               // N = 1024 -> 16 col tiles
  int lane = threadIdx.x & 31;
  int cl = lane & 15, hi = lane >> 4;

  v8f acc[4] = {};
  const half_t* A = ah + (size_t)m0 * CH;
#pragma unroll 2
  for (int k0 = 0; k0 < CH; k0 += 32) {
    v16h a = load_a16x32(A + k0, CH);
    v16h b[4];
#pragma unroll
    for (int t = 0; t < 4; ++t)
      b[t] = load_b32x16(wh + (size_t)(n0 + t * 16) * CH + k0, CH);
#pragma unroll
    for (int t = 0; t < 4; ++t)
      acc[t] = WMMA(a, b[t], acc[t]);
  }
#pragma unroll
  for (int t = 0; t < 4; ++t)
#pragma unroll
    for (int r = 0; r < 8; ++r) {
      int m = m0 + r + 8 * hi;
      int col = n0 + t * 16 + cl;
      out[(size_t)m * CH + col] = acc[t][r] + bias[col];
    }
}

extern "C" void kernel_launch(void* const* d_in, const int* in_sizes, int n_in,
                              void* d_out, int out_size, void* d_ws, size_t ws_size,
                              hipStream_t stream) {
  const float* x      = (const float*)d_in[0];   // [2, 2048, 1024]
  const float* w_qkv  = (const float*)d_in[1];   // [3072, 1024]
  const float* w_proj = (const float*)d_in[2];   // [1024, 1024]
  const float* b_proj = (const float*)d_in[3];   // [1024]
  const float* mask   = (const float*)d_in[4];   // [2048, 2048]
  float* out = (float*)d_out;                    // [2, 2048, 1024]

  half_t* ws = (half_t*)d_ws;
  size_t off = 0;
  half_t* xh  = ws + off; off += (size_t)4096 * 1024;    // x f16
  half_t* wqh = ws + off; off += (size_t)3072 * 1024;    // w_qkv f16
  half_t* wph = ws + off; off += (size_t)1024 * 1024;    // w_proj f16
  half_t* qh  = ws + off; off += (size_t)32 * 2048 * 64; // Q [B,H,N,D]
  half_t* kh  = ws + off; off += (size_t)32 * 2048 * 64; // K [B,H,N,D]
  half_t* vth = ws + off; off += (size_t)32 * 64 * 2048; // V^T [B,H,D,N]
  half_t* ah  = ws + off; off += (size_t)4096 * 1024;    // attn out f16
  (void)ws_size; (void)in_sizes; (void)n_in; (void)out_size;

  int n;
  n = 4096 * 1024;
  cvt_f32_to_f16<<<(n + 255) / 256, 256, 0, stream>>>(x, xh, n);
  n = 3072 * 1024;
  cvt_f32_to_f16<<<(n + 255) / 256, 256, 0, stream>>>(w_qkv, wqh, n);
  n = 1024 * 1024;
  cvt_f32_to_f16<<<(n + 255) / 256, 256, 0, stream>>>(w_proj, wph, n);

  gemm_qkv<<<1536, 256, 0, stream>>>(xh, wqh, qh, kh, vth);
  flash_attn<<<512, 256, 0, stream>>>(qh, kh, vth, mask, ah);
  gemm_proj<<<512, 256, 0, stream>>>(ah, wph, b_proj, out);
}
